// DqnPolicy4_30468497998096
// MI455X (gfx1250) — matrix operand order repaired and checked by reading, not compiled, not run on hardware
//
#include <hip/hip_runtime.h>
#include <hip/hip_bf16.h>

// ---------------------------------------------------------------------------
// MI455X / gfx1250, wave32. Matrix work through v_wmma_f32_16x16x32_f16.
// B weights pre-packed to fragment order (contiguous 32B per lane);
// A fragments = two contiguous 16B loads per lane; M=32 rows per wave.
// ---------------------------------------------------------------------------

typedef __attribute__((ext_vector_type(16))) _Float16 v16h;
typedef __attribute__((ext_vector_type(8)))  _Float16 v8h;
typedef __attribute__((ext_vector_type(8)))  float    v8f;

#define NI_CONST 100000
#define NS_CONST 100000

static __device__ __forceinline__ v8f wmma16(v16h a, v16h b, v8f c) {
  return __builtin_amdgcn_wmma_f32_16x16x32_f16(false, a, false, b, (short)0, c,
                                                false, false);
}

// ---- A fragment: 16x32 f16 (ISA 7.12.2). Lane's data = two contiguous
// 8-half runs: r[khalf*8 .. +7] and r[16+khalf*8 .. +7]  -> two 16B loads.
static __device__ __forceinline__ v16h load_A(const _Float16* __restrict__ base,
                                              int ld, int lane, int k0) {
  int row = lane & 15;
  int khalf = lane >> 4;
  const _Float16* r = base + row * ld + k0 + khalf * 8;
  v8h lo = *(const v8h*)(r);
  v8h hi = *(const v8h*)(r + 16);
  return __builtin_shufflevector(lo, hi, 0, 1, 2, 3, 4, 5, 6, 7, 8, 9, 10, 11,
                                 12, 13, 14, 15);
}

// ---- B fragment from packed weights: per (kt,nt) tile, 32 lanes x 16 halves
// contiguous. One 32-byte aligned load per lane.
static __device__ __forceinline__ v16h load_Bp(const _Float16* __restrict__ Wp,
                                               int numKt, int lane, int kt,
                                               int nt) {
  const v16h* p = (const v16h*)(Wp) + ((size_t)nt * numKt + kt) * 32 + lane;
  return *p;
}

// ---- C layout: lanes 0-15 col N=lane rows M=v..  lanes 16-31 rows M=8+v ----
static __device__ __forceinline__ void store_C(float* __restrict__ buf, int ld,
                                               int lane, int n0, v8f c) {
  int col = n0 + (lane & 15);
  int r0 = (lane >> 4) * 8;
#pragma unroll
  for (int v = 0; v < 8; ++v) buf[(r0 + v) * ld + col] = c[v];
}

// ---- LayerNorm (+optional ReLU) over D cols of a 16-row f32 LDS tile ------
static __device__ __forceinline__ void ln_act(const float* __restrict__ fb,
                                              _Float16* __restrict__ hb, int D,
                                              const float* __restrict__ g,
                                              const float* __restrict__ be,
                                              bool relu, int lane) {
  int row = lane & 15, part = lane >> 4;
  int half = D >> 1;
  float s = 0.f, s2 = 0.f;
  for (int j = part * half; j < part * half + half; ++j) {
    float v = fb[row * D + j];
    s += v;
    s2 += v * v;
  }
  s += __shfl_xor(s, 16);
  s2 += __shfl_xor(s2, 16);
  float mu = s / (float)D;
  float var = s2 / (float)D - mu * mu;
  float inv = rsqrtf(var + 1e-5f);
  for (int j = part * half; j < part * half + half; ++j) {
    float y = (fb[row * D + j] - mu) * inv * g[j] + be[j];
    if (relu) y = fmaxf(y, 0.f);
    hb[row * D + j] = (_Float16)y;
  }
}

// ===========================================================================
// Encoder + attention-score kernel: one wave per 32-row tile (2 sub-tiles
// sharing every B fragment). x[N,3] -> enc[N,128] f16, score s[N] f32.
// ===========================================================================
__global__ __launch_bounds__(32) void enc_kernel(
    const float* __restrict__ x,
    const float* __restrict__ w1, const float* __restrict__ b1,
    const float* __restrict__ g1, const float* __restrict__ be1,
    const _Float16* __restrict__ w2p, const float* __restrict__ b2,
    const float* __restrict__ g2, const float* __restrict__ be2,
    const _Float16* __restrict__ w3p, const float* __restrict__ b3,
    const float* __restrict__ g3, const float* __restrict__ be3,
    const _Float16* __restrict__ wa1p, const float* __restrict__ ba1,
    const float* __restrict__ wa2, const float* __restrict__ ba2,
    _Float16* __restrict__ encOut, float* __restrict__ scores, int N) {
  __shared__ float fbuf[32 * 128];
  __shared__ _Float16 hbuf[32 * 128];
  int lane = threadIdx.x;
  int row = lane & 15, part = lane >> 4;
  int col0 = lane & 15;
  int row0 = blockIdx.x * 32;

  // ---- layer 1 (K=3, scalar; ~0.5% of FLOPs) ----
#pragma unroll
  for (int s = 0; s < 2; ++s) {
    int gr = row0 + s * 16 + row;
    if (gr > N - 1) gr = N - 1;
    float x0 = x[gr * 3 + 0], x1 = x[gr * 3 + 1], x2 = x[gr * 3 + 2];
    for (int j = part * 32; j < part * 32 + 32; ++j) {
      fbuf[(s * 16 + row) * 64 + j] =
          x0 * w1[0 * 64 + j] + x1 * w1[1 * 64 + j] + x2 * w1[2 * 64 + j] + b1[j];
    }
  }
  __syncthreads();
#pragma unroll
  for (int s = 0; s < 2; ++s)
    ln_act(fbuf + s * 16 * 64, hbuf + s * 16 * 64, 64, g1, be1, true, lane);
  __syncthreads();

  // ---- layer 2: [32x64] @ [64x128] ----
  {
    v16h A0[2], A1[2];
#pragma unroll
    for (int s = 0; s < 2; ++s) {
      A0[s] = load_A(hbuf + s * 16 * 64, 64, lane, 0);
      A1[s] = load_A(hbuf + s * 16 * 64, 64, lane, 32);
    }
    for (int nt = 0; nt < 8; ++nt) {
      v16h B0 = load_Bp(w2p, 2, lane, 0, nt);
      v16h B1 = load_Bp(w2p, 2, lane, 1, nt);
      float bb = b2[nt * 16 + col0];
#pragma unroll
      for (int s = 0; s < 2; ++s) {
        v8f c = {};
        c = wmma16(A0[s], B0, c);
        c = wmma16(A1[s], B1, c);
#pragma unroll
        for (int v = 0; v < 8; ++v) c[v] += bb;
        store_C(fbuf + s * 16 * 128, 128, lane, nt * 16, c);
      }
    }
  }
  __syncthreads();
#pragma unroll
  for (int s = 0; s < 2; ++s)
    ln_act(fbuf + s * 16 * 128, hbuf + s * 16 * 128, 128, g2, be2, true, lane);
  __syncthreads();

  // ---- layer 3: [32x128] @ [128x128] ----
  {
    v16h A[2][4];
#pragma unroll
    for (int s = 0; s < 2; ++s)
#pragma unroll
      for (int k = 0; k < 4; ++k)
        A[s][k] = load_A(hbuf + s * 16 * 128, 128, lane, k * 32);
    for (int nt = 0; nt < 8; ++nt) {
      v16h B[4];
#pragma unroll
      for (int k = 0; k < 4; ++k) B[k] = load_Bp(w3p, 4, lane, k, nt);
      float bb = b3[nt * 16 + col0];
#pragma unroll
      for (int s = 0; s < 2; ++s) {
        v8f c = {};
#pragma unroll
        for (int k = 0; k < 4; ++k) c = wmma16(A[s][k], B[k], c);
#pragma unroll
        for (int v = 0; v < 8; ++v) c[v] += bb;
        store_C(fbuf + s * 16 * 128, 128, lane, nt * 16, c);
      }
    }
  }
  __syncthreads();
#pragma unroll
  for (int s = 0; s < 2; ++s)
    ln_act(fbuf + s * 16 * 128, hbuf + s * 16 * 128, 128, g3, be3, false, lane);
  // write enc tile (own lane's half-row; vectorized 16B chunks)
#pragma unroll
  for (int s = 0; s < 2; ++s) {
    int gr = row0 + s * 16 + row;
    if (gr < N) {
      const v8h* sp = (const v8h*)(hbuf + (s * 16 + row) * 128 + part * 64);
      v8h* dp = (v8h*)(encOut + (size_t)gr * 128 + part * 64);
#pragma unroll
      for (int q = 0; q < 8; ++q) dp[q] = sp[q];
    }
  }
  __syncthreads();

  // ---- attention score: tanh(enc@Wa1+ba1) @ wa2 + ba2 ----
  {
    v16h E[2][4];
#pragma unroll
    for (int s = 0; s < 2; ++s)
#pragma unroll
      for (int k = 0; k < 4; ++k)
        E[s][k] = load_A(hbuf + s * 16 * 128, 128, lane, k * 32);
    for (int nt = 0; nt < 4; ++nt) {
      v16h B[4];
#pragma unroll
      for (int k = 0; k < 4; ++k) B[k] = load_Bp(wa1p, 4, lane, k, nt);
      float bb = ba1[nt * 16 + col0];
      int col = nt * 16 + col0;
      int r0 = (lane >> 4) * 8;
#pragma unroll
      for (int s = 0; s < 2; ++s) {
        v8f c = {};
#pragma unroll
        for (int k = 0; k < 4; ++k) c = wmma16(E[s][k], B[k], c);
#pragma unroll
        for (int v = 0; v < 8; ++v)
          fbuf[(s * 16 + r0 + v) * 64 + col] = tanhf(c[v] + bb);
      }
    }
  }
  __syncthreads();
#pragma unroll
  for (int s = 0; s < 2; ++s) {
    float sd = 0.f;
    for (int j = part * 32; j < part * 32 + 32; ++j)
      sd += fbuf[(s * 16 + row) * 64 + j] * wa2[j];
    sd += __shfl_xor(sd, 16);
    int gr = row0 + s * 16 + row;
    if (part == 0 && gr < N) scores[gr] = sd + ba2[0];
  }
}

// ===========================================================================
// Q-net kernel: logit = relu(relu(c + enc@W1bot)@W2 + b2) @ w3 + b3
// ===========================================================================
__global__ __launch_bounds__(32) void qnet_kernel(
    const _Float16* __restrict__ enc, const float* __restrict__ cvec,
    const _Float16* __restrict__ w1bp, const _Float16* __restrict__ w2p,
    const float* __restrict__ b2, const float* __restrict__ w3,
    const float* __restrict__ b3, float* __restrict__ logits, int N) {
  __shared__ float fbuf[32 * 128];
  __shared__ _Float16 hbuf[32 * 128];
  int lane = threadIdx.x;
  int row = lane & 15, part = lane >> 4;
  int col0 = lane & 15;
  int row0 = blockIdx.x * 32;

  v16h E[2][4];
#pragma unroll
  for (int s = 0; s < 2; ++s)
#pragma unroll
    for (int k = 0; k < 4; ++k)
      E[s][k] = load_A(enc + (size_t)(row0 + s * 16) * 128, 128, lane, k * 32);

  // h1 = relu(c + enc @ W1bot)
  for (int nt = 0; nt < 8; ++nt) {
    v16h B[4];
#pragma unroll
    for (int k = 0; k < 4; ++k) B[k] = load_Bp(w1bp, 4, lane, k, nt);
    int col = nt * 16 + col0;
    float cc = cvec[col];
    int r0 = (lane >> 4) * 8;
#pragma unroll
    for (int s = 0; s < 2; ++s) {
      v8f c = {};
#pragma unroll
      for (int k = 0; k < 4; ++k) c = wmma16(E[s][k], B[k], c);
#pragma unroll
      for (int v = 0; v < 8; ++v)
        hbuf[(s * 16 + r0 + v) * 128 + col] = (_Float16)fmaxf(c[v] + cc, 0.f);
    }
  }
  __syncthreads();

  // h2 = relu(h1 @ W2 + b2)
  {
    v16h H[2][4];
#pragma unroll
    for (int s = 0; s < 2; ++s)
#pragma unroll
      for (int k = 0; k < 4; ++k)
        H[s][k] = load_A(hbuf + s * 16 * 128, 128, lane, k * 32);
    for (int nt = 0; nt < 4; ++nt) {
      v16h B[4];
#pragma unroll
      for (int k = 0; k < 4; ++k) B[k] = load_Bp(w2p, 4, lane, k, nt);
      int col = nt * 16 + col0;
      float bb = b2[col];
      int r0 = (lane >> 4) * 8;
#pragma unroll
      for (int s = 0; s < 2; ++s) {
        v8f c = {};
#pragma unroll
        for (int k = 0; k < 4; ++k) c = wmma16(H[s][k], B[k], c);
#pragma unroll
        for (int v = 0; v < 8; ++v)
          fbuf[(s * 16 + r0 + v) * 64 + col] = fmaxf(c[v] + bb, 0.f);
      }
    }
  }
  __syncthreads();
#pragma unroll
  for (int s = 0; s < 2; ++s) {
    float sd = 0.f;
    for (int j = part * 32; j < part * 32 + 32; ++j)
      sd += fbuf[(s * 16 + row) * 64 + j] * w3[j];
    sd += __shfl_xor(sd, 16);
    int gr = row0 + s * 16 + row;
    if (part == 0 && gr < N) logits[gr] = sd + b3[0];
  }
}

// ===========================================================================
// Weight pack: f32 row-major W[K][Nout] -> f16 fragment-order
// dst[(((nt*numKt)+kt)*32 + lane)*16 + j]  (lane's 16 halves contiguous)
// ===========================================================================
__global__ void pack_w_k(const float* __restrict__ src, _Float16* __restrict__ dst,
                         int K, int Nout) {
  int t = blockIdx.x * 256 + threadIdx.x;
  if (t >= K * Nout) return;
  int numKt = K >> 5;
  int j = t & 15;
  int lane = (t >> 4) & 31;
  int rest = t >> 9;
  int kt = rest % numKt;
  int nt = rest / numKt;
  int n = nt * 16 + (lane & 15);
  int k = kt * 32 + ((lane >> 4) << 3) + (j & 7) + ((j >> 3) << 4);
  dst[t] = (_Float16)src[k * Nout + n];
}

// ===========================================================================
// Small utility kernels (deterministic reductions, no float atomics)
// ===========================================================================
__global__ void reduce_max_k(const float* __restrict__ x, int n,
                             float* __restrict__ partial) {
  __shared__ float sm[256];
  float m = -3.4e38f;
  for (int i = blockIdx.x * 256 + threadIdx.x; i < n; i += gridDim.x * 256)
    m = fmaxf(m, x[i]);
  sm[threadIdx.x] = m;
  __syncthreads();
  for (int st = 128; st > 0; st >>= 1) {
    if (threadIdx.x < st) sm[threadIdx.x] = fmaxf(sm[threadIdx.x], sm[threadIdx.x + st]);
    __syncthreads();
  }
  if (threadIdx.x == 0) partial[blockIdx.x] = sm[0];
}

__global__ void reduce_sum_k(const float* __restrict__ x, int n,
                             float* __restrict__ partial) {
  __shared__ float sm[256];
  float a = 0.f;
  for (int i = blockIdx.x * 256 + threadIdx.x; i < n; i += gridDim.x * 256) a += x[i];
  sm[threadIdx.x] = a;
  __syncthreads();
  for (int st = 128; st > 0; st >>= 1) {
    if (threadIdx.x < st) sm[threadIdx.x] += sm[threadIdx.x + st];
    __syncthreads();
  }
  if (threadIdx.x == 0) partial[blockIdx.x] = sm[0];
}

__global__ void reduce_sumexp_k(const float* __restrict__ x, int n,
                                const float* __restrict__ gmax,
                                float* __restrict__ partial) {
  __shared__ float sm[256];
  float m = gmax[0];
  float a = 0.f;
  for (int i = blockIdx.x * 256 + threadIdx.x; i < n; i += gridDim.x * 256)
    a += expf(x[i] - m);
  sm[threadIdx.x] = a;
  __syncthreads();
  for (int st = 128; st > 0; st >>= 1) {
    if (threadIdx.x < st) sm[threadIdx.x] += sm[threadIdx.x + st];
    __syncthreads();
  }
  if (threadIdx.x == 0) partial[blockIdx.x] = sm[0];
}

// agg partials: block b accumulates rows b, b+G, ... (fixed order, deterministic)
__global__ void attn_agg_part_k(const _Float16* __restrict__ enc,
                                const float* __restrict__ s,
                                const float* __restrict__ gmax, int n,
                                float* __restrict__ part) {
  int c = threadIdx.x;  // 0..127
  float m = gmax[0];
  float acc = 0.f;
  for (int i = blockIdx.x; i < n; i += gridDim.x)
    acc += expf(s[i] - m) * (float)enc[(size_t)i * 128 + c];
  part[blockIdx.x * 128 + c] = acc;
}

__global__ void attn_agg_final_k(const float* __restrict__ part, int nb,
                                 float* __restrict__ agg) {
  int c = threadIdx.x;
  float acc = 0.f;
  for (int b = 0; b < nb; ++b) acc += part[b * 128 + c];
  agg[c] = acc;
}

// fused MLP (256->512->256) + per-qnet constant c = fused@W1top + b1
__global__ __launch_bounds__(256) void fused_k(
    const float* __restrict__ aggI, const float* __restrict__ sumI,
    const float* __restrict__ aggS, const float* __restrict__ sumS,
    const float* __restrict__ fw1, const float* __restrict__ fb1,
    const float* __restrict__ fw2, const float* __restrict__ fb2,
    const float* __restrict__ iqw1, const float* __restrict__ iqb1,
    const float* __restrict__ sqw1, const float* __restrict__ sqb1,
    float* __restrict__ cI, float* __restrict__ cS) {
  __shared__ float h0[256], h1[512], h2[256];
  int t = threadIdx.x;
  if (t < 128)
    h0[t] = aggI[t] / sumI[0];
  else
    h0[t] = aggS[t - 128] / sumS[0];
  __syncthreads();
  for (int o = t; o < 512; o += 256) {
    float a = fb1[o];
    for (int k = 0; k < 256; ++k) a += h0[k] * fw1[k * 512 + o];
    h1[o] = fmaxf(a, 0.f);
  }
  __syncthreads();
  {
    float a = fb2[t];
    for (int k = 0; k < 512; ++k) a += h1[k] * fw2[k * 256 + t];
    h2[t] = fmaxf(a, 0.f);
  }
  __syncthreads();
  if (t < 128) {
    float a = iqb1[t];
    for (int k = 0; k < 256; ++k) a += h2[k] * iqw1[k * 128 + t];
    cI[t] = a;
  } else {
    int o = t - 128;
    float a = sqb1[o];
    for (int k = 0; k < 256; ++k) a += h2[k] * sqw1[k * 128 + o];
    cS[o] = a;
  }
}

__global__ void softmax_out_k(const float* __restrict__ l,
                              const float* __restrict__ gmax,
                              const float* __restrict__ gsum,
                              float* __restrict__ out, int n) {
  int i = blockIdx.x * blockDim.x + threadIdx.x;
  if (i < n) out[i] = expf(l[i] - gmax[0]) / gsum[0];
}

// ===========================================================================
// Host driver
// ===========================================================================
extern "C" void kernel_launch(void* const* d_in, const int* in_sizes, int n_in,
                              void* d_out, int out_size, void* d_ws, size_t ws_size,
                              hipStream_t stream) {
  (void)in_sizes; (void)n_in; (void)out_size; (void)ws_size;
  const int NI = NI_CONST, NS = NS_CONST;

  // ---- input pointer map (setup_inputs dict order, depth-first) ----
  const float* itemX = (const float*)d_in[0];
  const float* spaceX = (const float*)d_in[1];
  const float* ie_w1 = (const float*)d_in[2];  const float* ie_b1 = (const float*)d_in[3];
  const float* ie_g1 = (const float*)d_in[4];  const float* ie_e1 = (const float*)d_in[5];
  const float* ie_w2 = (const float*)d_in[6];  const float* ie_b2 = (const float*)d_in[7];
  const float* ie_g2 = (const float*)d_in[8];  const float* ie_e2 = (const float*)d_in[9];
  const float* ie_w3 = (const float*)d_in[10]; const float* ie_b3 = (const float*)d_in[11];
  const float* ie_g3 = (const float*)d_in[12]; const float* ie_e3 = (const float*)d_in[13];
  const float* se_w1 = (const float*)d_in[14]; const float* se_b1 = (const float*)d_in[15];
  const float* se_g1 = (const float*)d_in[16]; const float* se_e1 = (const float*)d_in[17];
  const float* se_w2 = (const float*)d_in[18]; const float* se_b2 = (const float*)d_in[19];
  const float* se_g2 = (const float*)d_in[20]; const float* se_e2 = (const float*)d_in[21];
  const float* se_w3 = (const float*)d_in[22]; const float* se_b3 = (const float*)d_in[23];
  const float* se_g3 = (const float*)d_in[24]; const float* se_e3 = (const float*)d_in[25];
  const float* ia_w1 = (const float*)d_in[26]; const float* ia_b1 = (const float*)d_in[27];
  const float* ia_w2 = (const float*)d_in[28]; const float* ia_b2 = (const float*)d_in[29];
  const float* sa_w1 = (const float*)d_in[30]; const float* sa_b1 = (const float*)d_in[31];
  const float* sa_w2 = (const float*)d_in[32]; const float* sa_b2 = (const float*)d_in[33];
  const float* fu_w1 = (const float*)d_in[34]; const float* fu_b1 = (const float*)d_in[35];
  const float* fu_w2 = (const float*)d_in[36]; const float* fu_b2 = (const float*)d_in[37];
  const float* iq_w1 = (const float*)d_in[38]; const float* iq_b1 = (const float*)d_in[39];
  const float* iq_w2 = (const float*)d_in[40]; const float* iq_b2 = (const float*)d_in[41];
  const float* iq_w3 = (const float*)d_in[42]; const float* iq_b3 = (const float*)d_in[43];
  const float* sq_w1 = (const float*)d_in[44]; const float* sq_b1 = (const float*)d_in[45];
  const float* sq_w2 = (const float*)d_in[46]; const float* sq_b2 = (const float*)d_in[47];
  const float* sq_w3 = (const float*)d_in[48]; const float* sq_b3 = (const float*)d_in[49];

  // ---- workspace layout ----
  size_t off = 0;
  auto give = [&](size_t bytes) -> void* {
    void* p = (char*)d_ws + off;
    off = (off + bytes + 255) & ~(size_t)255;
    return p;
  };
  _Float16* encI = (_Float16*)give((size_t)NI * 128 * 2);
  _Float16* encS = (_Float16*)give((size_t)NS * 128 * 2);
  _Float16* p_ie2 = (_Float16*)give(64 * 128 * 2);
  _Float16* p_ie3 = (_Float16*)give(128 * 128 * 2);
  _Float16* p_ia1 = (_Float16*)give(128 * 64 * 2);
  _Float16* p_se2 = (_Float16*)give(64 * 128 * 2);
  _Float16* p_se3 = (_Float16*)give(128 * 128 * 2);
  _Float16* p_sa1 = (_Float16*)give(128 * 64 * 2);
  _Float16* p_iq1b = (_Float16*)give(128 * 128 * 2);
  _Float16* p_iq2 = (_Float16*)give(128 * 64 * 2);
  _Float16* p_sq1b = (_Float16*)give(128 * 128 * 2);
  _Float16* p_sq2 = (_Float16*)give(128 * 64 * 2);
  float* scoresI = (float*)give((size_t)NI * 4);
  float* scoresS = (float*)give((size_t)NS * 4);
  float* logitsI = (float*)give((size_t)NI * 4);
  float* logitsS = (float*)give((size_t)NS * 4);
  float* partial = (float*)give(256 * 4);
  float* aggpart = (float*)give(512 * 128 * 4);
  float* aggI = (float*)give(128 * 4);
  float* aggS = (float*)give(128 * 4);
  float* cI = (float*)give(128 * 4);
  float* cS = (float*)give(128 * 4);
  float* scal = (float*)give(8 * 4);
  float *smaxI = scal + 0, *ssumI = scal + 1, *smaxS = scal + 2, *ssumS = scal + 3;
  float *lmaxI = scal + 4, *lsumI = scal + 5, *lmaxS = scal + 6, *lsumS = scal + 7;

  auto pack = [&](const float* s, _Float16* d, int K, int Nout) {
    pack_w_k<<<(K * Nout + 255) / 256, 256, 0, stream>>>(s, d, K, Nout);
  };
  // ---- weight f32->f16 fragment-order packing ----
  pack(ie_w2, p_ie2, 64, 128);
  pack(ie_w3, p_ie3, 128, 128);
  pack(ia_w1, p_ia1, 128, 64);
  pack(se_w2, p_se2, 64, 128);
  pack(se_w3, p_se3, 128, 128);
  pack(sa_w1, p_sa1, 128, 64);
  pack(iq_w1 + 256 * 128, p_iq1b, 128, 128);  // bottom 128 rows of [384,128]
  pack(iq_w2, p_iq2, 128, 64);
  pack(sq_w1 + 256 * 128, p_sq1b, 128, 128);
  pack(sq_w2, p_sq2, 128, 64);

  // ---- encoders (one wave per 32-row tile) ----
  int tilesI = (NI + 31) / 32, tilesS = (NS + 31) / 32;
  enc_kernel<<<tilesI, 32, 0, stream>>>(itemX, ie_w1, ie_b1, ie_g1, ie_e1, p_ie2,
                                        ie_b2, ie_g2, ie_e2, p_ie3, ie_b3, ie_g3,
                                        ie_e3, p_ia1, ia_b1, ia_w2, ia_b2, encI,
                                        scoresI, NI);
  enc_kernel<<<tilesS, 32, 0, stream>>>(spaceX, se_w1, se_b1, se_g1, se_e1, p_se2,
                                        se_b2, se_g2, se_e2, p_se3, se_b3, se_g3,
                                        se_e3, p_sa1, sa_b1, sa_w2, sa_b2, encS,
                                        scoresS, NS);

  // ---- attention softmax stats + aggregation (deterministic) ----
  reduce_max_k<<<256, 256, 0, stream>>>(scoresI, NI, partial);
  reduce_max_k<<<1, 256, 0, stream>>>(partial, 256, smaxI);
  reduce_sumexp_k<<<256, 256, 0, stream>>>(scoresI, NI, smaxI, partial);
  reduce_sum_k<<<1, 256, 0, stream>>>(partial, 256, ssumI);
  attn_agg_part_k<<<512, 128, 0, stream>>>(encI, scoresI, smaxI, NI, aggpart);
  attn_agg_final_k<<<1, 128, 0, stream>>>(aggpart, 512, aggI);

  reduce_max_k<<<256, 256, 0, stream>>>(scoresS, NS, partial);
  reduce_max_k<<<1, 256, 0, stream>>>(partial, 256, smaxS);
  reduce_sumexp_k<<<256, 256, 0, stream>>>(scoresS, NS, smaxS, partial);
  reduce_sum_k<<<1, 256, 0, stream>>>(partial, 256, ssumS);
  attn_agg_part_k<<<512, 128, 0, stream>>>(encS, scoresS, smaxS, NS, aggpart);
  attn_agg_final_k<<<1, 128, 0, stream>>>(aggpart, 512, aggS);

  // ---- fused MLP + q-net constant vectors ----
  fused_k<<<1, 256, 0, stream>>>(aggI, ssumI, aggS, ssumS, fu_w1, fu_b1, fu_w2,
                                 fu_b2, iq_w1, iq_b1, sq_w1, sq_b1, cI, cS);

  // ---- q-nets ----
  qnet_kernel<<<tilesI, 32, 0, stream>>>(encI, cI, p_iq1b, p_iq2, iq_b2, iq_w3,
                                         iq_b3, logitsI, NI);
  qnet_kernel<<<tilesS, 32, 0, stream>>>(encS, cS, p_sq1b, p_sq2, sq_b2, sq_w3,
                                         sq_b3, logitsS, NS);

  // ---- final softmaxes -> d_out ----
  float* out = (float*)d_out;
  reduce_max_k<<<256, 256, 0, stream>>>(logitsI, NI, partial);
  reduce_max_k<<<1, 256, 0, stream>>>(partial, 256, lmaxI);
  reduce_sumexp_k<<<256, 256, 0, stream>>>(logitsI, NI, lmaxI, partial);
  reduce_sum_k<<<1, 256, 0, stream>>>(partial, 256, lsumI);
  softmax_out_k<<<(NI + 255) / 256, 256, 0, stream>>>(logitsI, lmaxI, lsumI, out, NI);

  reduce_max_k<<<256, 256, 0, stream>>>(logitsS, NS, partial);
  reduce_max_k<<<1, 256, 0, stream>>>(partial, 256, lmaxS);
  reduce_sumexp_k<<<256, 256, 0, stream>>>(logitsS, NS, lmaxS, partial);
  reduce_sum_k<<<1, 256, 0, stream>>>(partial, 256, lsumS);
  softmax_out_k<<<(NS + 255) / 256, 256, 0, stream>>>(logitsS, lmaxS, lsumS,
                                                      out + NI, NS);
}
